// SwinTransformerBlock_11287174054485
// MI455X (gfx1250) — compile-verified, matrix operand order checked by reading, not run on hardware
//
#include <hip/hip_runtime.h>
#include <hip/hip_bf16.h>
#include <math.h>

typedef __attribute__((ext_vector_type(16))) _Float16 v16h;
typedef __attribute__((ext_vector_type(8)))  float    v8f;

// ---- geometry ----
#define BB   4
#define HH_  128
#define WW_  256
#define CC   192
#define NHD  6
#define HEAD 32
#define NTOK 64          // 8x8 window
#define HID  768
#define NWIN 2048        // 4 * 16 * 32

// padded LDS strides (in halves) to de-phase banks
#define LDX 200          // rows of 192
#define LDQ 40           // rows of 32
#define LDV 72           // rows of 64

// -----------------------------------------------------------------------------
// Fragment loader: CDNA5 16-bit A/B layout for V_WMMA_F32_16X16X32_F16.
// tile = pointer to element (row0, k0) of a 16 x 32 sub-tile, row-major,
// ld = row stride in halves (must be even).
// -----------------------------------------------------------------------------
__device__ __forceinline__ v16h frag_ld(const _Float16* tile, int ld) {
  const int lane = threadIdx.x & 31;
  const unsigned* p = (const unsigned*)(tile + (lane & 15) * ld);
  const int ko = (lane >> 4) * 4;          // uint index offset (8 halves)
  union { unsigned u[8]; v16h h; } f;
#pragma unroll
  for (int i = 0; i < 4; ++i) {
    f.u[i]     = p[ko + i];                // K = koff + 2i, +1
    f.u[i + 4] = p[ko + 8 + i];            // K = 16 + koff + 2i, +1
  }
  return f.h;
}

__device__ __forceinline__ v8f wmma_step(v16h a, v16h b, v8f c) {
  return __builtin_amdgcn_wmma_f32_16x16x32_f16(false, a, false, b, (short)0, c,
                                                false, false);
}

// -----------------------------------------------------------------------------
// Kernel 0: weights f32->f16, gather relative-position bias to [NH][64][64]
// -----------------------------------------------------------------------------
__global__ __launch_bounds__(256) void prep_kernel(
    const float* __restrict__ qkv_w, const float* __restrict__ proj_w,
    const float* __restrict__ fc1_w, const float* __restrict__ fc2_w,
    const float* __restrict__ rpb_table, const int* __restrict__ rel_idx,
    _Float16* __restrict__ qkv16, _Float16* __restrict__ proj16,
    _Float16* __restrict__ fc116, _Float16* __restrict__ fc216,
    float* __restrict__ biasTab) {
  const int n1 = 3 * CC * CC, n2 = CC * CC, n3 = HID * CC, n4 = CC * HID,
            n5 = NHD * NTOK * NTOK;
  const int total = n1 + n2 + n3 + n4 + n5;
  for (int idx = blockIdx.x * blockDim.x + threadIdx.x; idx < total;
       idx += gridDim.x * blockDim.x) {
    int j = idx;
    if (j < n1) { qkv16[j] = (_Float16)qkv_w[j]; continue; }
    j -= n1;
    if (j < n2) { proj16[j] = (_Float16)proj_w[j]; continue; }
    j -= n2;
    if (j < n3) { fc116[j] = (_Float16)fc1_w[j]; continue; }
    j -= n3;
    if (j < n4) { fc216[j] = (_Float16)fc2_w[j]; continue; }
    j -= n4;
    const int h = j >> 12;          // /4096
    const int mn = j & 4095;        // m*64+n
    biasTab[j] = rpb_table[rel_idx[mn] * NHD + h];
  }
}

// -----------------------------------------------------------------------------
// Kernel 1: LN1 + shifted-window + QKV + cosine attention + proj + residual
// one workgroup (8 wave32) per window
// -----------------------------------------------------------------------------
__global__ __launch_bounds__(256) void attn_kernel(
    const float* __restrict__ x, const float* __restrict__ n1g,
    const float* __restrict__ n1b, const _Float16* __restrict__ qkvw,
    const float* __restrict__ qkvb, const float* __restrict__ logit_scale,
    const float* __restrict__ biasTab, const _Float16* __restrict__ projw,
    const float* __restrict__ projb, float* __restrict__ x1) {
  extern __shared__ char smem[];
  _Float16* xn   = (_Float16*)smem;                 // 64 x LDX  (LN out / attn out)
  _Float16* qbuf = xn + NTOK * LDX;                 // [h][tok][d]  6*64 x LDQ
  _Float16* kbuf = qbuf + NHD * NTOK * LDQ;         // [h][tok][d]
  _Float16* vbuf = kbuf + NHD * NTOK * LDQ;         // [h][d][tok]  6*32 x LDV
  float*    S32  = (float*)(vbuf + NHD * HEAD * LDV);   // 64 x 64
  _Float16* P16  = (_Float16*)(S32 + NTOK * NTOK);      // 64 x LDV
  float*    qinv = (float*)(P16 + NTOK * LDV);          // 6*64
  float*    kinv = qinv + NHD * NTOK;                   // 6*64

  const int tid  = threadIdx.x;
  const int lane = tid & 31;
  const int wave = tid >> 5;
  const int wid  = blockIdx.x;
  const int bi   = wid >> 9;         // / (16*32)
  const int wy   = (wid & 511) >> 5;
  const int wx   = wid & 31;

  // ---- LayerNorm1 over shifted-window gather (4 threads / token) ----
  {
    const int t = tid >> 2, part = tid & 3;
    const int hh = (wy * 8 + (t >> 3) + 4) & (HH_ - 1);
    const int ww = (wx * 8 + (t & 7) + 4) & (WW_ - 1);
    const float* row = x + (((size_t)bi * HH_ + hh) * WW_ + ww) * CC;
    float v[48], s = 0.f, ss = 0.f;
#pragma unroll
    for (int c = 0; c < 48; ++c) {
      float f = row[part * 48 + c];
      v[c] = f; s += f; ss += f * f;
    }
    s  += __shfl_xor(s, 1);  ss += __shfl_xor(ss, 1);
    s  += __shfl_xor(s, 2);  ss += __shfl_xor(ss, 2);
    const float mu = s * (1.f / CC);
    const float rstd = rsqrtf(ss * (1.f / CC) - mu * mu + 1e-5f);
#pragma unroll
    for (int c = 0; c < 48; ++c) {
      const int ch = part * 48 + c;
      xn[t * LDX + ch] = (_Float16)((v[c] - mu) * rstd * n1g[ch] + n1b[ch]);
    }
  }
  __syncthreads();

  // ---- QKV GEMM: 64x576 = 4x36 tiles of 16x16, K=192 ----
  for (int t = wave; t < 144; t += 8) {
    const int mt = t / 36, nt = t % 36;
    const float b0 = qkvb[nt * 16 + (lane & 15)];
    v8f acc = {b0, b0, b0, b0, b0, b0, b0, b0};
#pragma unroll
    for (int ks = 0; ks < 6; ++ks) {
      v16h a = frag_ld(xn + mt * 16 * LDX + ks * 32, LDX);
      v16h b = frag_ld(qkvw + (size_t)nt * 16 * CC + ks * 32, CC);
      acc = wmma_step(a, b, acc);
    }
    // scatter into q / k / v(T).  which / h / d-base depend only on nt
    // (tiles never straddle a q/k/v or head boundary) -> wave-uniform.
    const int which = nt / 12;               // 0=q, 1=k, 2=v
    const int ntl   = nt % 12;
    const int h     = ntl >> 1;              // head (uniform)
    const int d     = ((ntl & 1) << 4) + (lane & 15);
    const int mhi   = (lane >> 4) << 3;
    if (which == 2) {
#pragma unroll
      for (int r = 0; r < 8; ++r)
        vbuf[(h * HEAD + d) * LDV + mt * 16 + r + mhi] = (_Float16)acc[r];
    } else {
      _Float16* qk = which ? kbuf : qbuf;
#pragma unroll
      for (int r = 0; r < 8; ++r)
        qk[(h * NTOK + mt * 16 + r + mhi) * LDQ + d] = (_Float16)acc[r];
    }
  }
  __syncthreads();

  // ---- inverse L2 norms for q and k rows ----
  for (int task = tid; task < 2 * NHD * NTOK; task += 256) {
    const bool isQ = task < NHD * NTOK;
    const int r = isQ ? task : task - NHD * NTOK;
    const _Float16* p = (isQ ? qbuf : kbuf) + r * LDQ;
    float s = 0.f;
#pragma unroll
    for (int d = 0; d < HEAD; ++d) { float f = (float)p[d]; s += f * f; }
    (isQ ? qinv : kinv)[r] = 1.f / fmaxf(sqrtf(s), 1e-12f);
  }
  __syncthreads();

  // ---- per-head attention ----
  const float lmax = logf(100.0f);
  for (int h = 0; h < NHD; ++h) {
    const float hscale = __expf(fminf(logit_scale[h], lmax));
    // logits: 16 tiles (2 per wave), K = HEAD = 32 -> one WMMA
#pragma unroll
    for (int tt = 0; tt < 2; ++tt) {
      const int t = wave * 2 + tt;
      const int mt = t >> 2, nt = t & 3;
      v16h a = frag_ld(qbuf + (h * NTOK + mt * 16) * LDQ, LDQ);
      v16h b = frag_ld(kbuf + (h * NTOK + nt * 16) * LDQ, LDQ);
      v8f acc = {};
      acc = wmma_step(a, b, acc);
      const int nloc = nt * 16 + (lane & 15);
      const float ks = kinv[h * NTOK + nloc] * hscale;
      const int mhi = (lane >> 4) << 3;
#pragma unroll
      for (int r = 0; r < 8; ++r) {
        const int mloc = mt * 16 + r + mhi;
        S32[mloc * NTOK + nloc] =
            acc[r] * qinv[h * NTOK + mloc] * ks +
            biasTab[(h * NTOK + mloc) * NTOK + nloc];
      }
    }
    __syncthreads();
    // softmax: 4 threads per row, 16 columns each, shuffle reductions
    {
      const int row = tid >> 2, part = tid & 3;
      float vloc[16];
      float mx = -1e30f;
#pragma unroll
      for (int j = 0; j < 16; ++j) {
        vloc[j] = S32[row * NTOK + part * 16 + j];
        mx = fmaxf(mx, vloc[j]);
      }
      mx = fmaxf(mx, __shfl_xor(mx, 1));
      mx = fmaxf(mx, __shfl_xor(mx, 2));
      float sum = 0.f;
#pragma unroll
      for (int j = 0; j < 16; ++j) {
        vloc[j] = __expf(vloc[j] - mx);
        sum += vloc[j];
      }
      sum += __shfl_xor(sum, 1);
      sum += __shfl_xor(sum, 2);
      const float inv = 1.f / sum;
#pragma unroll
      for (int j = 0; j < 16; ++j)
        P16[row * LDV + part * 16 + j] = (_Float16)(vloc[j] * inv);
    }
    __syncthreads();
    // P @ V : 8 tiles, one per wave, K = 64 -> 2 WMMA steps
    {
      const int mt = wave >> 1, dt = wave & 1;
      v8f acc = {};
#pragma unroll
      for (int ks = 0; ks < 2; ++ks) {
        v16h a = frag_ld(P16 + mt * 16 * LDV + ks * 32, LDV);
        v16h b = frag_ld(vbuf + (h * HEAD + dt * 16) * LDV + ks * 32, LDV);
        acc = wmma_step(a, b, acc);
      }
      const int c = h * HEAD + dt * 16 + (lane & 15);
      const int mhi = (lane >> 4) << 3;
#pragma unroll
      for (int r = 0; r < 8; ++r)
        xn[(mt * 16 + r + mhi) * LDX + c] = (_Float16)acc[r];
    }
    __syncthreads();
  }

  // ---- proj GEMM (64x192, K=192) + reverse roll scatter + residual ----
  for (int t = wave; t < 48; t += 8) {
    const int mt = t / 12, ct = t % 12;
    const int cg = ct * 16 + (lane & 15);
    const float b0 = projb[cg];
    v8f acc = {b0, b0, b0, b0, b0, b0, b0, b0};
#pragma unroll
    for (int ks = 0; ks < 6; ++ks) {
      v16h a = frag_ld(xn + mt * 16 * LDX + ks * 32, LDX);
      v16h b = frag_ld(projw + (size_t)ct * 16 * CC + ks * 32, CC);
      acc = wmma_step(a, b, acc);
    }
    const int mhi = (lane >> 4) << 3;
#pragma unroll
    for (int r = 0; r < 8; ++r) {
      const int m = mt * 16 + r + mhi;
      const int hh = (wy * 8 + (m >> 3) + 4) & (HH_ - 1);
      const int ww = (wx * 8 + (m & 7) + 4) & (WW_ - 1);
      const size_t g = (((size_t)bi * HH_ + hh) * WW_ + ww) * CC + cg;
      x1[g] = x[g] + acc[r];
    }
  }
}

// -----------------------------------------------------------------------------
// Kernel 2: LN2 + fc1 + exact GELU + fc2 + residual.  64-token tiles,
// HID chunked 4 x 192 so hidden activations stay in LDS.
// -----------------------------------------------------------------------------
__global__ __launch_bounds__(256) void mlp_kernel(
    const float* __restrict__ x1, const float* __restrict__ n2g,
    const float* __restrict__ n2b, const _Float16* __restrict__ w1,
    const float* __restrict__ b1, const _Float16* __restrict__ w2,
    const float* __restrict__ b2, float* __restrict__ out) {
  __shared__ _Float16 A16[NTOK * LDX];
  __shared__ _Float16 Hc[NTOK * LDX];
  const int tid = threadIdx.x, lane = tid & 31, wave = tid >> 5;
  const size_t row0 = (size_t)blockIdx.x * NTOK;

  // LN2
  {
    const int t = tid >> 2, part = tid & 3;
    const float* row = x1 + (row0 + t) * CC;
    float v[48], s = 0.f, ss = 0.f;
#pragma unroll
    for (int c = 0; c < 48; ++c) {
      float f = row[part * 48 + c];
      v[c] = f; s += f; ss += f * f;
    }
    s  += __shfl_xor(s, 1);  ss += __shfl_xor(ss, 1);
    s  += __shfl_xor(s, 2);  ss += __shfl_xor(ss, 2);
    const float mu = s * (1.f / CC);
    const float rstd = rsqrtf(ss * (1.f / CC) - mu * mu + 1e-5f);
#pragma unroll
    for (int c = 0; c < 48; ++c) {
      const int ch = part * 48 + c;
      A16[t * LDX + ch] = (_Float16)((v[c] - mu) * rstd * n2g[ch] + n2b[ch]);
    }
  }
  __syncthreads();

  // output accumulators: 6 tiles of 16x16 per wave (4 x 12 tiles total)
  v8f acc[6];
  int mt[6], ct[6];
#pragma unroll
  for (int ti = 0; ti < 6; ++ti) {
    const int t = ti * 8 + wave;
    mt[ti] = t / 12; ct[ti] = t % 12;
    const float b0 = b2[ct[ti] * 16 + (lane & 15)];
    acc[ti] = (v8f){b0, b0, b0, b0, b0, b0, b0, b0};
  }

  for (int kc = 0; kc < 4; ++kc) {
    __syncthreads();   // prior Hc readers done
    // fc1 chunk: hidden cols [kc*192, kc*192+192)
#pragma unroll
    for (int ti = 0; ti < 6; ++ti) {
      const int t = ti * 8 + wave;
      const int fmt = t / 12, fnt = t % 12;
      const int ng = kc * CC + fnt * 16 + (lane & 15);
      const float b0 = b1[ng];
      v8f ha = {b0, b0, b0, b0, b0, b0, b0, b0};
#pragma unroll
      for (int ks = 0; ks < 6; ++ks) {
        v16h a = frag_ld(A16 + fmt * 16 * LDX + ks * 32, LDX);
        v16h b = frag_ld(w1 + ((size_t)kc * CC + fnt * 16) * CC + ks * 32, CC);
        ha = wmma_step(a, b, ha);
      }
      const int mhi = (lane >> 4) << 3;
#pragma unroll
      for (int r = 0; r < 8; ++r) {
        const float v0 = ha[r];
        const float ge = 0.5f * v0 * (1.f + erff(v0 * 0.70710678118f));
        Hc[(fmt * 16 + r + mhi) * LDX + fnt * 16 + (lane & 15)] = (_Float16)ge;
      }
    }
    __syncthreads();
    // fc2 accumulate over this chunk's K = 192
#pragma unroll
    for (int ti = 0; ti < 6; ++ti) {
#pragma unroll
      for (int ks = 0; ks < 6; ++ks) {
        v16h a = frag_ld(Hc + mt[ti] * 16 * LDX + ks * 32, LDX);
        v16h b = frag_ld(w2 + (size_t)ct[ti] * 16 * HID + kc * CC + ks * 32, HID);
        acc[ti] = wmma_step(a, b, acc[ti]);
      }
    }
  }

  // final residual write
  const int mhi = (lane >> 4) << 3;
#pragma unroll
  for (int ti = 0; ti < 6; ++ti) {
    const int cg = ct[ti] * 16 + (lane & 15);
#pragma unroll
    for (int r = 0; r < 8; ++r) {
      const int m = mt[ti] * 16 + r + mhi;
      const size_t g = (row0 + m) * CC + cg;
      out[g] = x1[g] + acc[ti][r];
    }
  }
}

// -----------------------------------------------------------------------------
extern "C" void kernel_launch(void* const* d_in, const int* in_sizes, int n_in,
                              void* d_out, int out_size, void* d_ws,
                              size_t ws_size, hipStream_t stream) {
  const float* x        = (const float*)d_in[0];
  const float* n1g      = (const float*)d_in[1];
  const float* n1b      = (const float*)d_in[2];
  const float* qkv_w    = (const float*)d_in[3];
  const float* qkv_b    = (const float*)d_in[4];
  const float* lscale   = (const float*)d_in[5];
  const float* rpb      = (const float*)d_in[6];
  const float* proj_w   = (const float*)d_in[7];
  const float* proj_b   = (const float*)d_in[8];
  const float* n2g      = (const float*)d_in[9];
  const float* n2b      = (const float*)d_in[10];
  const float* fc1_w    = (const float*)d_in[11];
  const float* fc1_b    = (const float*)d_in[12];
  const float* fc2_w    = (const float*)d_in[13];
  const float* fc2_b    = (const float*)d_in[14];
  const int*   rel_idx  = (const int*)d_in[15];

  char* ws = (char*)d_ws;
  size_t off = 0;
  float* x1 = (float*)(ws + off);        off += (size_t)BB * HH_ * WW_ * CC * 4;
  _Float16* qkv16  = (_Float16*)(ws + off); off += (size_t)3 * CC * CC * 2;
  _Float16* proj16 = (_Float16*)(ws + off); off += (size_t)CC * CC * 2;
  _Float16* fc116  = (_Float16*)(ws + off); off += (size_t)HID * CC * 2;
  _Float16* fc216  = (_Float16*)(ws + off); off += (size_t)CC * HID * 2;
  float* biasTab   = (float*)(ws + off);

  prep_kernel<<<512, 256, 0, stream>>>(qkv_w, proj_w, fc1_w, fc2_w, rpb,
                                       rel_idx, qkv16, proj16, fc116, fc216,
                                       biasTab);

  const size_t smem1 =
      (NTOK * LDX + 2 * NHD * NTOK * LDQ + NHD * HEAD * LDV + NTOK * LDV) * 2 +
      NTOK * NTOK * 4 + 2 * NHD * NTOK * 4;
  attn_kernel<<<NWIN, 256, smem1, stream>>>(x, n1g, n1b, qkv16, qkv_b, lscale,
                                            biasTab, proj16, proj_b, x1);

  mlp_kernel<<<NWIN, 256, 0, stream>>>(x1, n2g, n2b, fc116, fc1_b, fc216,
                                       fc2_b, (float*)d_out);
}
